// MultiHeadedSelfAttention_13477607375378
// MI455X (gfx1250) — matrix operand order repaired and checked by reading, not compiled, not run on hardware
//
#include <hip/hip_runtime.h>
#include <hip/hip_bf16.h>

static constexpr int NB = 4, NS = 1024, ND = 1024, NH = 16, NHD = 64;

typedef __attribute__((ext_vector_type(16))) __bf16 bf16x16;
typedef __attribute__((ext_vector_type(8)))  __bf16 bf16x8;
typedef __attribute__((ext_vector_type(8)))  float  f32x8;

#if __has_builtin(__builtin_amdgcn_tensor_load_to_lds) && __has_builtin(__builtin_amdgcn_s_wait_tensorcnt)
#define HAVE_TDM 1
#else
#define HAVE_TDM 0
#endif

// ---------------- bf16 helpers: native converts (gfx1250 is bf16-native) ----------------
static __device__ __forceinline__ __bf16 f2bf(float f) { return (__bf16)f; }
static __device__ __forceinline__ float  bf2f(__bf16 b) { return (float)b; }

// ---------------- WMMA fragment loaders ----------------
static __device__ __forceinline__ bf16x16 frag_ld2(const __bf16* p0, const __bf16* p1) {
  bf16x8 lo = *(const bf16x8*)p0;
  bf16x8 hi = *(const bf16x8*)p1;
  bf16x16 f;
#pragma unroll
  for (int e = 0; e < 8; ++e) { f[e] = lo[e]; f[e + 8] = hi[e]; }
  return f;
}
// A fragment (16x32): line = this lane's row base, kb = ((lane>>4)<<3).
#define LOAD_A(line, koff, kb) frag_ld2((line) + (koff) + (kb), (line) + (koff) + 16 + (kb))
// B fragment (32x16): line = this lane's column (row of B^T), kh = ((lane>>4)<<4).
#define LOAD_B(line, koff, kh) frag_ld2((line) + (koff) + ((kh) >> 1), (line) + (koff) + 16 + ((kh) >> 1))

static __device__ __forceinline__ f32x8 wmma_bf16(bf16x16 a, bf16x16 b, f32x8 c) {
  return __builtin_amdgcn_wmma_f32_16x16x32_bf16(false, a, false, b, (short)0, c, false, false);
}

#if HAVE_TDM
typedef __attribute__((ext_vector_type(4))) unsigned u32x4;
typedef __attribute__((ext_vector_type(8))) int      i32x8;
typedef __attribute__((ext_vector_type(4))) int      i32x4;

// Issue a 2D-tile TDM load (D# per CDNA5 ISA §8): data_size=8B units, count=1, type=2.
// tile_d0/tensor_d0 in 8-byte units; stride_units = row stride in 8-byte units.
static __device__ __forceinline__ void tdm_load_2d(unsigned lds_addr, unsigned long long gaddr,
                                                   unsigned tile_d0, unsigned tile_d1,
                                                   unsigned long long stride_units) {
  u32x4 g0;
  g0[0] = 1u;                                               // count=1 (valid), user mode
  g0[1] = lds_addr;                                         // LDS byte address
  g0[2] = (unsigned)(gaddr & 0xffffffffu);                  // global_addr[31:0]
  g0[3] = (unsigned)((gaddr >> 32) & 0x1ffffffu) | (2u << 30);  // global_addr[56:32] | type=2
  i32x8 g1;
  const unsigned tensor_d0 = tile_d0, tensor_d1 = tile_d1;  // interior tile: no OOB clipping
  g1[0] = (int)(3u << 16);                                  // wg_mask=0, data_size=3 (8B)
  g1[1] = (int)((tensor_d0 & 0xffffu) << 16);               // tensor_dim0[15:0] @bits63:48
  g1[2] = (int)(((tensor_d0 >> 16) & 0xffffu) | ((tensor_d1 & 0xffffu) << 16));
  g1[3] = (int)(((tensor_d1 >> 16) & 0xffffu) | ((tile_d0 & 0xffffu) << 16));  // tile_dim0
  g1[4] = (int)(tile_d1 & 0xffffu);                         // tile_dim1 (tile_dim2=0 -> 2D)
  g1[5] = (int)(unsigned)(stride_units & 0xffffffffu);      // tensor_dim0_stride[31:0]
  g1[6] = (int)(unsigned)((stride_units >> 32) & 0xffffu);  // tensor_dim0_stride[47:32]
  g1[7] = 0;
  i32x4 gz = {};
#if __clang_major__ >= 23
  i32x8 gz8 = {};
  __builtin_amdgcn_tensor_load_to_lds(g0, g1, gz, gz, gz8, 0);
#else
  __builtin_amdgcn_tensor_load_to_lds(g0, g1, gz, gz, 0);
#endif
}
#endif  // HAVE_TDM

// ---------------- LayerNorm + bf16 cast ----------------
__global__ __launch_bounds__(256)
void ln_kernel(const float* __restrict__ x, const float* __restrict__ gamma,
               const float* __restrict__ beta, __bf16* __restrict__ xn) {
  const int row = blockIdx.x;                 // 0..B*S-1
  const int t = threadIdx.x;
  const float* xr = x + (size_t)row * ND;
  __shared__ float red[256];
  float v4[4]; float s = 0.f;
#pragma unroll
  for (int i = 0; i < 4; ++i) { v4[i] = xr[t + 256 * i]; s += v4[i]; }
  red[t] = s; __syncthreads();
  for (int o = 128; o > 0; o >>= 1) { if (t < o) red[t] += red[t + o]; __syncthreads(); }
  const float mu = red[0] * (1.f / ND);
  __syncthreads();
  float q = 0.f;
#pragma unroll
  for (int i = 0; i < 4; ++i) { float d = v4[i] - mu; q += d * d; }
  red[t] = q; __syncthreads();
  for (int o = 128; o > 0; o >>= 1) { if (t < o) red[t] += red[t + o]; __syncthreads(); }
  const float rstd = rsqrtf(red[0] * (1.f / ND) + 1e-5f);
#pragma unroll
  for (int i = 0; i < 4; ++i) {
    int c = t + 256 * i;
    xn[(size_t)row * ND + c] = f2bf((v4[i] - mu) * rstd * gamma[c] + beta[c]);
  }
}

// ---------------- Sinusoidal positional encodings (bf16) ----------------
__global__ __launch_bounds__(256)
void pe_kernel(__bf16* __restrict__ pe) {
  const int t = blockIdx.x * 256 + threadIdx.x;   // over S*D
  const int s = t / ND, d = t % ND;
  const float inv = __expf(-(float)(d & ~1) * (9.210340371976184f / (float)ND));
  const float ang = (float)s * inv;
  pe[t] = f2bf((d & 1) ? __cosf(ang) : __sinf(ang));
}

// ---------------- fp32 -> bf16 transposed weight convert ----------------
__global__ __launch_bounds__(256)
void cvtT_kernel(const float* __restrict__ src, __bf16* __restrict__ dst, int K, int N) {
  int t = blockIdx.x * 256 + threadIdx.x;
  if (t < K * N) {
    int k = t / N, n = t % N;                    // coalesced read along n
    dst[(size_t)n * K + k] = f2bf(src[t]);
  }
}

// ---------------- GEMM: C[M,N] = A[M,K] @ W[K,N], W given transposed ----------------
// No LDS, no barriers: both operands are L2-resident streams of contiguous 16B lines.
template <int MODE>
__global__ __launch_bounds__(128)
void gemm_kernel(const __bf16* __restrict__ A, const __bf16* __restrict__ Wt,
                 const float* __restrict__ bias, const float* __restrict__ uvec,
                 const float* __restrict__ vvec,
                 __bf16* __restrict__ out0, __bf16* __restrict__ out1,
                 float* __restrict__ outf, int M, int N, int K) {
  const int tid = threadIdx.x, lane = tid & 31;
  const int wave = __builtin_amdgcn_readfirstlane(tid >> 5);   // wave-uniform -> SALU control
  const int wm = (wave >> 1) * 32, wn = (wave & 1) * 64;
  const int m0 = blockIdx.x * 64, n0 = blockIdx.y * 128;
  const int col = lane & 15, kb = (lane >> 4) << 3, kh = (lane >> 4) << 4;
  const int rhalf = (lane >> 4) * 8;
  f32x8 acc[2][4] = {};
  const __bf16* aline0 = A + (size_t)(m0 + wm + col) * K;
  const __bf16* aline1 = aline0 + (size_t)16 * K;
  const __bf16* bline[4];
#pragma unroll
  for (int j = 0; j < 4; ++j) bline[j] = Wt + (size_t)(n0 + wn + j * 16 + col) * K;
  for (int k0 = 0; k0 < K; k0 += 32) {
    if (k0 + 64 < K) {
      __builtin_prefetch(aline0 + k0 + 64, 0, 0);
      __builtin_prefetch(bline[0] + k0 + 64, 0, 0);
    }
    bf16x16 a0 = LOAD_A(aline0, k0, kb);
    bf16x16 a1 = LOAD_A(aline1, k0, kb);
#pragma unroll
    for (int j = 0; j < 4; ++j) {
      bf16x16 bj = LOAD_B(bline[j], k0, kh);
      acc[0][j] = wmma_bf16(a0, bj, acc[0][j]);
      acc[1][j] = wmma_bf16(a1, bj, acc[1][j]);
    }
  }
#pragma unroll
  for (int i = 0; i < 2; ++i)
#pragma unroll
    for (int j = 0; j < 4; ++j) {
      const int nn = n0 + wn + j * 16 + col;
      const float bia = (MODE == 2) ? 0.f : bias[nn];
#pragma unroll
      for (int r = 0; r < 8; ++r) {
        const int mm = m0 + wm + i * 16 + rhalf + r;
        const float val = acc[i][j][r] + bia;
        if (MODE == 1) {
          const size_t off = (size_t)mm * N + nn;
          out0[off] = f2bf(val + uvec[nn]); out1[off] = f2bf(val + vvec[nn]);
        } else if (MODE == 3) {
          outf[(size_t)mm * N + nn] = val;
        } else if (MODE == 4) {
          out0[(size_t)nn * M + mm] = f2bf(val);      // V stored transposed [D, B*S]
        } else {
          out0[(size_t)mm * N + nn] = f2bf(val);
        }
      }
    }
}

// ---------------- Fused rel-pos flash attention ----------------
// Phase 1: Braw band in LDS (stride S+1 + zero gap -> rel-shift is one contiguous read).
// Phase 2: flash over 32-key blocks. K and V^T blocks are DMA'd into double-buffered LDS
//          by the Tensor Data Mover (one wave issues, TENSORcnt-tracked, next block's DMA
//          overlaps current block's WMMAs) -- removes the 4x duplicate global fetches the
//          four waves would otherwise issue. Falls back to direct L2 streams without TDM.
__global__ __launch_bounds__(128)
void attn_kernel(const __bf16* __restrict__ qu, const __bf16* __restrict__ qv,
                 const __bf16* __restrict__ kmat, const __bf16* __restrict__ vT,
                 const __bf16* __restrict__ pmat, __bf16* __restrict__ O) {
  __shared__ __align__(16) __bf16 sBraw[66 * 1025];     // rows 0..64 + dummy row 65
  __shared__ __align__(16) __bf16 sP[4][16 * 40];       // per-wave P staging [row][k 0..31]
#if HAVE_TDM
  __shared__ __align__(16) __bf16 sK[2][32 * 64];       // K block: [key][hd]
  __shared__ __align__(16) __bf16 sV[2][64 * 32];       // V^T block: [d][key]
#endif
  const int tid = threadIdx.x, lane = tid & 31;
  const int wave = __builtin_amdgcn_readfirstlane(tid >> 5);   // wave-uniform -> SALU control
  const int iblk = blockIdx.x, bh = blockIdx.y;
  const int b = bh / NH, h = bh % NH;
  const int i0 = iblk * 64;
  const int col = lane & 15, kb = (lane >> 4) << 3, kh = (lane >> 4) << 4;
  const int rhalf = (lane >> 4) * 8;
  const size_t headoff = (size_t)h * NHD;

  // -------- Phase 1: Braw band --------
  for (int rt = 0; rt < 5; ++rt) {
    int row = i0 + rt * 16 + col; if (row > NS - 1) row = NS - 1;
    const __bf16* qline = qv + ((size_t)(b * NS + row)) * ND + headoff;
    bf16x16 a0 = LOAD_A(qline, 0, kb);
    bf16x16 a1 = LOAD_A(qline, 32, kb);
    for (int ct = wave; ct < 64; ct += 4) {
      const __bf16* pline = pmat + (size_t)(ct * 16 + col) * ND + headoff;
      bf16x16 b0 = LOAD_B(pline, 0, kh);
      bf16x16 b1 = LOAD_B(pline, 32, kh);
      f32x8 acc = {};
      acc = wmma_bf16(a0, b0, acc);
      acc = wmma_bf16(a1, b1, acc);
#pragma unroll
      for (int r = 0; r < 8; ++r) {                      // branchless: excess rows -> dummy 65
        int R = rt * 16 + rhalf + r;
        int Rs = (R < 65) ? R : 65;
        sBraw[Rs * 1025 + ct * 16 + col] = f2bf(acc[r]);
      }
    }
  }
  for (int R = tid; R < 65; R += 128) sBraw[R * 1025 + 1024] = f2bf(0.f);  // rel-shift zero gap

  // -------- Phase 2: flash attention (wave -> 16-row query tile) --------
  const int irow_base = i0 + wave * 16;
  const __bf16* quline = qu + (size_t)(b * NS + irow_base + col) * ND + headoff;
  bf16x16 aq0 = LOAD_A(quline, 0, kb);
  bf16x16 aq1 = LOAD_A(quline, 32, kb);
  f32x8 accO[4] = {};
  float mrow[8], lrow[8];
#pragma unroll
  for (int r = 0; r < 8; ++r) { mrow[r] = -3.0e38f; lrow[r] = 0.f; }
  const __bf16* kbase = kmat + (size_t)b * NS * ND + headoff;
  const __bf16* vbase0 = vT + (size_t)(h * NHD) * (NB * NS) + (size_t)b * NS;  // d-line 0
#if !HAVE_TDM
  const __bf16* vline[4];
#pragma unroll
  for (int n = 0; n < 4; ++n) vline[n] = vbase0 + (size_t)(n * 16 + col) * (NB * NS);
#endif
  // BD band lookup: R*1025 + (S-1-(i0+R)) + j  ==  (S-1-i0) + R*1024 + j
  const int bdbase = (NS - 1 - i0) + col;

#if HAVE_TDM
  if (wave == 0) {   // preload block 0 (K: 32x128B rows, V^T: 64x64B rows; 8B units)
    tdm_load_2d((unsigned)(size_t)&sK[0][0], (unsigned long long)(size_t)kbase,
                16, 32, (unsigned long long)(ND * 2 / 8));
    tdm_load_2d((unsigned)(size_t)&sV[0][0], (unsigned long long)(size_t)vbase0,
                8, 64, (unsigned long long)((size_t)NB * NS * 2 / 8));
  }
#endif
  __syncthreads();   // phase-1 band visible (TDM path re-syncs below)

  for (int kblk = 0; kblk < NS; kblk += 32) {
#if HAVE_TDM
    const int cur = (kblk >> 5) & 1;
    if (wave == 0) {
      if (kblk + 32 < NS) {     // overlap next block's DMA with this block's WMMAs
        tdm_load_2d((unsigned)(size_t)&sK[cur ^ 1][0],
                    (unsigned long long)(size_t)(kbase + (size_t)(kblk + 32) * ND),
                    16, 32, (unsigned long long)(ND * 2 / 8));
        tdm_load_2d((unsigned)(size_t)&sV[cur ^ 1][0],
                    (unsigned long long)(size_t)(vbase0 + kblk + 32),
                    8, 64, (unsigned long long)((size_t)NB * NS * 2 / 8));
        __builtin_amdgcn_s_wait_tensorcnt(2);   // cur's 2 TDMs done; next's in flight
      } else {
        __builtin_amdgcn_s_wait_tensorcnt(0);
      }
    }
    __syncthreads();                            // cur block visible to all waves
#else
    if (kblk + 32 < NS) {
      __builtin_prefetch(kbase + (size_t)(kblk + 32 + col) * ND, 0, 0);
      __builtin_prefetch(vline[0] + kblk + 32, 0, 0);
    }
#endif
    // scores for 32 keys: AC via WMMA + BD via band lookup
    float sc[2][8];
#pragma unroll
    for (int c = 0; c < 2; ++c) {
#if HAVE_TDM
      const __bf16* kline = &sK[cur][(size_t)(c * 16 + col) * 64];
#else
      const __bf16* kline = kbase + (size_t)(kblk + c * 16 + col) * ND;
#endif
      bf16x16 bk0 = LOAD_B(kline, 0, kh);
      bf16x16 bk1 = LOAD_B(kline, 32, kh);
      f32x8 s = {};
      s = wmma_bf16(aq0, bk0, s);
      s = wmma_bf16(aq1, bk1, s);
#pragma unroll
      for (int r = 0; r < 8; ++r) {
        const int R = wave * 16 + rhalf + r;
        const float bd = bf2f(sBraw[bdbase + R * 1024 + kblk + c * 16]);
        sc[c][r] = (s[r] + bd) * 0.03125f;   // 1/sqrt(D)
      }
    }
    // online softmax (row stats shared by the 16 lanes holding that row)
    float pr[2][8];
#pragma unroll
    for (int r = 0; r < 8; ++r) {
      float mx = fmaxf(sc[0][r], sc[1][r]);
#pragma unroll
      for (int msk = 1; msk < 16; msk <<= 1) mx = fmaxf(mx, __shfl_xor(mx, msk, 32));
      const float mnew = fmaxf(mrow[r], mx);
      const float alpha = __expf(mrow[r] - mnew);
      const float p0 = __expf(sc[0][r] - mnew);
      const float p1 = __expf(sc[1][r] - mnew);
      float ls = p0 + p1;
#pragma unroll
      for (int msk = 1; msk < 16; msk <<= 1) ls += __shfl_xor(ls, msk, 32);
      lrow[r] = lrow[r] * alpha + ls;
      mrow[r] = mnew;
      pr[0][r] = p0; pr[1][r] = p1;
#pragma unroll
      for (int n = 0; n < 4; ++n) accO[n][r] *= alpha;
    }
    // P tile -> per-wave LDS staging (D-layout out, A-layout back in).
#pragma unroll
    for (int c = 0; c < 2; ++c)
#pragma unroll
      for (int r = 0; r < 8; ++r)
        sP[wave][(rhalf + r) * 40 + c * 16 + col] = f2bf(pr[c][r]);
    asm volatile("" ::: "memory");
    const __bf16* pline2 = &sP[wave][(size_t)col * 40];
    bf16x16 aP = LOAD_A(pline2, 0, kb);
#pragma unroll
    for (int n = 0; n < 4; ++n) {
#if HAVE_TDM
      const __bf16* vl = &sV[cur][(size_t)(n * 16 + col) * 32];
      bf16x16 bV = LOAD_B(vl, 0, kh);
#else
      bf16x16 bV = LOAD_B(vline[n], kblk, kh);
#endif
      accO[n] = wmma_bf16(aP, bV, accO[n]);
    }
#if HAVE_TDM
    __syncthreads();            // all waves done with cur before its buffer is re-filled
#endif
  }
  // normalize and store O (bf16)
#pragma unroll
  for (int r = 0; r < 8; ++r) {
    const int i = irow_base + rhalf + r;
    const float inv = 1.f / lrow[r];
    __bf16* oline = O + (size_t)(b * NS + i) * ND + headoff;
#pragma unroll
    for (int n = 0; n < 4; ++n) oline[n * 16 + col] = f2bf(accO[n][r] * inv);
  }
}

// ---------------- host launcher ----------------
extern "C" void kernel_launch(void* const* d_in, const int* in_sizes, int n_in,
                              void* d_out, int out_size, void* d_ws, size_t ws_size,
                              hipStream_t stream) {
  (void)in_sizes; (void)n_in; (void)out_size; (void)ws_size;
  const float* x     = (const float*)d_in[0];
  const float* Wq    = (const float*)d_in[1];
  const float* bq    = (const float*)d_in[2];
  const float* Wk    = (const float*)d_in[3];
  const float* bk    = (const float*)d_in[4];
  const float* Wv    = (const float*)d_in[5];
  const float* bv    = (const float*)d_in[6];
  const float* Wpos  = (const float*)d_in[7];
  const float* Wf    = (const float*)d_in[8];
  const float* bfin  = (const float*)d_in[9];
  const float* uvec  = (const float*)d_in[10];
  const float* vvec  = (const float*)d_in[11];
  const float* gamma = (const float*)d_in[12];
  const float* beta  = (const float*)d_in[13];
  float* out = (float*)d_out;

  char* ws = (char*)d_ws;
  size_t off = 0;
  auto alloc = [&](size_t bytes) -> char* {
    char* p = ws + off;
    off += (bytes + 255) & ~(size_t)255;
    return p;
  };
  const size_t ACT = (size_t)NB * NS * ND * 2;   // 8 MB bf16 activation
  const size_t WB  = (size_t)ND * ND * 2;        // 2 MB bf16 weight
  __bf16* xnB    = (__bf16*)alloc(ACT);
  __bf16* peB    = (__bf16*)alloc((size_t)NS * ND * 2);
  __bf16* WqT    = (__bf16*)alloc(WB);
  __bf16* WkT    = (__bf16*)alloc(WB);
  __bf16* WvT    = (__bf16*)alloc(WB);
  __bf16* WposT  = (__bf16*)alloc(WB);
  __bf16* WfT    = (__bf16*)alloc(WB);
  __bf16* quB    = (__bf16*)alloc(ACT);
  __bf16* qvB    = (__bf16*)alloc(ACT);
  __bf16* kB     = (__bf16*)alloc(ACT);
  __bf16* vTB    = (__bf16*)alloc(ACT);
  __bf16* pB     = (__bf16*)alloc((size_t)NS * ND * 2);
  __bf16* OB     = (__bf16*)alloc(ACT);

  const int M = NB * NS;                          // 4096
  ln_kernel<<<dim3(M), dim3(256), 0, stream>>>(x, gamma, beta, xnB);
  pe_kernel<<<dim3(NS * ND / 256), dim3(256), 0, stream>>>(peB);
  const int WN = ND * ND;
  cvtT_kernel<<<dim3((WN + 255) / 256), dim3(256), 0, stream>>>(Wq,   WqT,   ND, ND);
  cvtT_kernel<<<dim3((WN + 255) / 256), dim3(256), 0, stream>>>(Wk,   WkT,   ND, ND);
  cvtT_kernel<<<dim3((WN + 255) / 256), dim3(256), 0, stream>>>(Wv,   WvT,   ND, ND);
  cvtT_kernel<<<dim3((WN + 255) / 256), dim3(256), 0, stream>>>(Wpos, WposT, ND, ND);
  cvtT_kernel<<<dim3((WN + 255) / 256), dim3(256), 0, stream>>>(Wf,   WfT,   ND, ND);

  gemm_kernel<1><<<dim3(M / 64, ND / 128), dim3(128), 0, stream>>>(
      xnB, WqT, bq, uvec, vvec, quB, qvB, nullptr, M, ND, ND);
  gemm_kernel<0><<<dim3(M / 64, ND / 128), dim3(128), 0, stream>>>(
      xnB, WkT, bk, nullptr, nullptr, kB, nullptr, nullptr, M, ND, ND);
  gemm_kernel<4><<<dim3(M / 64, ND / 128), dim3(128), 0, stream>>>(
      xnB, WvT, bv, nullptr, nullptr, vTB, nullptr, nullptr, M, ND, ND);
  gemm_kernel<2><<<dim3(NS / 64, ND / 128), dim3(128), 0, stream>>>(
      peB, WposT, nullptr, nullptr, nullptr, pB, nullptr, nullptr, NS, ND, ND);

  attn_kernel<<<dim3(NS / 64, NB * NH), dim3(128), 0, stream>>>(quB, qvB, kB, vTB, pB, OB);

  gemm_kernel<3><<<dim3(M / 64, ND / 128), dim3(128), 0, stream>>>(
      OB, WfT, bfin, nullptr, nullptr, nullptr, nullptr, out, M, ND, ND);
}
